// FGDN_43164421324860
// MI455X (gfx1250) — matrix operand order repaired and checked by reading, not compiled, not run on hardware
//
#include <hip/hip_runtime.h>
#include <math.h>

// ---------------- CDNA5 WMMA / TDM types ----------------
typedef __bf16 v16bf __attribute__((ext_vector_type(16)));
typedef float  v8f   __attribute__((ext_vector_type(8)));
typedef unsigned int v8u __attribute__((ext_vector_type(8)));
typedef unsigned int u32x4 __attribute__((ext_vector_type(4)));
typedef int i32x4 __attribute__((ext_vector_type(4)));
typedef int i32x8 __attribute__((ext_vector_type(8)));

#define NNODES 100000
#define INF    128
#define DF     512
#define NG     64
#define NC     10
#define EPSBN  1e-5f

__device__ __forceinline__ unsigned short f2bf(float f) {
    unsigned int u = __float_as_uint(f);
    u = (u + 0x7FFFu + ((u >> 16) & 1u)) >> 16;   // round-to-nearest-even
    return (unsigned short)u;
}
__device__ __forceinline__ float bf2f(unsigned short h) {
    return __uint_as_float(((unsigned int)h) << 16);
}

// ---------------------------------------------------------------------------
// TDM: DMA a (128 rows x 64 cols) bf16 tile of A into LDS.
// global_addr points at tile start; tensor_dim1 = rows remaining so the
// hardware zero-fills out-of-bounds rows (handles the M tail for free).
// ---------------------------------------------------------------------------
__device__ __forceinline__ void tdm_load_a_tile(const unsigned short* gptr,
                                                unsigned lds_off,
                                                int K, int rows_left)
{
    unsigned long long ga = (unsigned long long)(size_t)gptr;
    u32x4 g0;
    g0[0] = 1u;                                        // count=1, gather off
    g0[1] = lds_off;                                   // lds_addr [63:32]
    g0[2] = (unsigned)(ga & 0xffffffffu);              // global_addr lo
    g0[3] = (unsigned)((ga >> 32) & 0x01ffffffu)       // global_addr hi (57b)
          | (2u << 30);                                // type = 2 ("image")
    i32x8 g1;
    g1[0] = 0x00010000;                                // data_size=1 (2 bytes)
    g1[1] = (int)(((unsigned)K & 0xffffu) << 16);      // tensor_dim0 lo16
    g1[2] = (int)((((unsigned)K) >> 16)                // tensor_dim0 hi16
          | (((unsigned)rows_left & 0xffffu) << 16));  // tensor_dim1 lo16
    g1[3] = (int)((((unsigned)rows_left) >> 16)        // tensor_dim1 hi16
          | (64u << 16));                              // tile_dim0 = 64
    g1[4] = 128;                                       // tile_dim1=128, tile_dim2=0
    g1[5] = K;                                         // tensor_dim0_stride lo32
    g1[6] = 0;                                         // stride0 hi / stride1 lo
    g1[7] = 0;
    i32x4 gz; gz[0] = 0; gz[1] = 0; gz[2] = 0; gz[3] = 0;
#if defined(__clang_major__) && (__clang_major__ >= 23)
    i32x8 gz8;
    for (int i = 0; i < 8; ++i) gz8[i] = 0;
    __builtin_amdgcn_tensor_load_to_lds(g0, g1, gz, gz, gz8, 0);
#else
    __builtin_amdgcn_tensor_load_to_lds(g0, g1, gz, gz, 0);
#endif
}

// ---------------------------------------------------------------------------
// Kernel 1: column mean/var stats of x (N x 128) + fp32 -> bf16 conversion
// ---------------------------------------------------------------------------
__global__ void __launch_bounds__(256)
stats_cvt_kernel(const float* __restrict__ x, unsigned short* __restrict__ xbf,
                 float* __restrict__ ssum, float* __restrict__ sss, int M)
{
    __shared__ float ps[256], ps2[256];
    const int t = threadIdx.x;
    const int c = t & 127;
    const int ro = t >> 7;                 // 0 or 1
    const int base = blockIdx.x * 128;
    float s = 0.f, s2 = 0.f;
    #pragma unroll 4
    for (int i = 0; i < 64; ++i) {
        int row = base + ro + 2 * i;
        if (row < M) {
            float v = x[(size_t)row * INF + c];
            s += v; s2 += v * v;
            xbf[(size_t)row * INF + c] = f2bf(v);
        }
    }
    ps[t] = s; ps2[t] = s2;
    __syncthreads();
    if (t < 128) {
        atomicAdd(&ssum[c], ps[t] + ps[t + 128]);
        atomicAdd(&sss[c],  ps2[t] + ps2[t + 128]);
    }
}

// ---------------------------------------------------------------------------
// Kernel 2a: fold BN1 into w0, pack B into WMMA layout ([col][k/2] bf16 pairs)
// ---------------------------------------------------------------------------
__global__ void __launch_bounds__(128)
prep_w0_kernel(const float* __restrict__ w0, const float* __restrict__ b0,
               const float* __restrict__ g, const float* __restrict__ be,
               const float* __restrict__ ssum, const float* __restrict__ sss,
               unsigned short* __restrict__ wpk, float* __restrict__ biasp,
               float invN)
{
    __shared__ float red[128];
    const int k = threadIdx.x;
    const int j = blockIdx.x;
    float mean = ssum[k] * invN;
    float var  = sss[k] * invN - mean * mean;
    float sc = g[k] * rsqrtf(var + EPSBN);
    float sh = be[k] - mean * sc;
    float wv = w0[(size_t)k * DF + j];
    wpk[(size_t)j * INF + k] = f2bf(sc * wv);
    red[k] = sh * wv;
    __syncthreads();
    for (int s = 64; s > 0; s >>= 1) {
        if (k < s) red[k] += red[k + s];
        __syncthreads();
    }
    if (k == 0) biasp[j] = b0[j] + red[0];
}

// ---------------------------------------------------------------------------
// Kernel 2b: plain bf16 pack of w1 (no BN fold)
// ---------------------------------------------------------------------------
__global__ void __launch_bounds__(256)
prep_plain_kernel(const float* __restrict__ w, unsigned short* __restrict__ wpk)
{
    const int j = blockIdx.x;
    for (int k = threadIdx.x; k < DF; k += 256)
        wpk[(size_t)j * DF + k] = f2bf(w[(size_t)k * DF + j]);
}

// ---------------------------------------------------------------------------
// Kernel 2c: fold BN3 (stats of previous relu output) into w2/w3 + pack
// ---------------------------------------------------------------------------
__global__ void __launch_bounds__(256)
prep_bn_kernel(const float* __restrict__ w, const float* __restrict__ b,
               const float* __restrict__ g, const float* __restrict__ be,
               const float* __restrict__ ssum, const float* __restrict__ sss,
               unsigned short* __restrict__ wpk, float* __restrict__ biasp,
               float invN)
{
    __shared__ float red[256];
    const int t = threadIdx.x;
    const int j = blockIdx.x;
    float part = 0.f;
    for (int k = t; k < DF; k += 256) {
        float mean = ssum[k] * invN;
        float var  = sss[k] * invN - mean * mean;
        float sc = g[k] * rsqrtf(var + EPSBN);
        float sh = be[k] - mean * sc;
        float wv = w[(size_t)k * DF + j];
        wpk[(size_t)j * DF + k] = f2bf(sc * wv);
        part += sh * wv;
    }
    red[t] = part;
    __syncthreads();
    for (int s = 128; s > 0; s >>= 1) {
        if (t < s) red[t] += red[t + s];
        __syncthreads();
    }
    if (t == 0) biasp[j] = b[j] + red[0];
}

// ---------------------------------------------------------------------------
// Kernel 3: main bf16 WMMA GEMM with TDM double-buffered A staging (K-step 64)
// Out = relu(A @ Bpk + bias), optional per-column sum/sumsq stats.
// A: M x K bf16 row-major.  Bpk: packed B [col][k/2] uint (bf16 pair).
// Block = 256 thr (8 waves), tile 128x128.  Epilogue: bf16 tile staged in the
// (dead) A double-buffer LDS, streamed out as coalesced b128 stores.
// ---------------------------------------------------------------------------
__global__ void __launch_bounds__(256)
gemm_bf16_kernel(const unsigned short* __restrict__ A,
                 const unsigned short* __restrict__ Bpk,
                 const float* __restrict__ bias,
                 unsigned short* __restrict__ Out,
                 int M, int K,
                 float* __restrict__ ssum, float* __restrict__ sss)
{
    __shared__ __align__(16) unsigned short lA[2][128 * 64];  // 2 x 16 KB
    __shared__ float lcs[128], lcss[128];

    const int t    = threadIdx.x;
    const int wave = t >> 5;
    const int lane = t & 31;
    const int hf   = lane >> 4;       // lane half (0/1)
    const int lr   = lane & 15;
    const int m0 = blockIdx.x * 128;
    const int n0 = blockIdx.y * 128;
    const int wm = (wave & 3) * 32;   // wave row offset inside block tile
    const int wn = (wave >> 2) * 64;  // wave col offset inside block tile
    const int rows_left = M - m0;     // TDM zero-fills rows >= rows_left

    if (t < 128) { lcs[t] = 0.f; lcss[t] = 0.f; }

    v8f acc[2][4];
    #pragma unroll
    for (int i = 0; i < 2; ++i)
        #pragma unroll
        for (int j = 0; j < 4; ++j)
            acc[i][j] = v8f{0.f,0.f,0.f,0.f,0.f,0.f,0.f,0.f};

    const unsigned int* Bu = (const unsigned int*)Bpk;
    const int Kp = K >> 1;            // uints per column of packed B
    const unsigned short* Abase = A + (size_t)m0 * K;
    const int nsteps = K >> 6;        // K-step = 64

    // prologue: DMA tile 0 into buffer 0 (wave 0 issues; TENSORcnt tracks)
    if (wave == 0) {
        tdm_load_a_tile(Abase, (unsigned)(size_t)&lA[0][0], K, rows_left);
        __builtin_amdgcn_s_wait_tensorcnt(0);
    }
    __syncthreads();

    for (int s = 0; s < nsteps; ++s) {
        const int k0  = s << 6;
        const int buf = s & 1;
        // overlap: DMA next 128x64 tile into the other buffer
        if (wave == 0 && (s + 1) < nsteps) {
            tdm_load_a_tile(Abase + (k0 + 64),
                            (unsigned)(size_t)&lA[buf ^ 1][0], K, rows_left);
        }

        #pragma unroll
        for (int kk = 0; kk < 64; kk += 32) {
            // A fragments (16x32 bf16), ISA 16-bit A layout:
            // lanes 0-15: K 0..7 (dw0-3), 16..23 (dw4-7); lanes 16-31: +8
            v16bf af[2];
            #pragma unroll
            for (int mt = 0; mt < 2; ++mt) {
                const int row = wm + mt * 16 + lr;
                const uint4 c0 = *(const uint4*)(&lA[buf][0] + row * 64 + kk + hf * 8);
                const uint4 c1 = *(const uint4*)(&lA[buf][0] + row * 64 + kk + 16 + hf * 8);
                v8u u;
                u[0]=c0.x; u[1]=c0.y; u[2]=c0.z; u[3]=c0.w;
                u[4]=c1.x; u[5]=c1.y; u[6]=c1.z; u[7]=c1.w;
                af[mt] = __builtin_bit_cast(v16bf, u);
            }
            // B fragments straight from (L2-resident) packed weights + WMMA
            #pragma unroll
            for (int nt = 0; nt < 4; ++nt) {
                const int col = n0 + wn + nt * 16 + lr;
                const size_t base = (size_t)col * Kp + ((k0 + kk) >> 1) + hf * 4;
                const uint4 b0 = *(const uint4*)(Bu + base);
                const uint4 b1 = *(const uint4*)(Bu + base + 8);
                v8u u;
                u[0]=b0.x; u[1]=b0.y; u[2]=b0.z; u[3]=b0.w;
                u[4]=b1.x; u[5]=b1.y; u[6]=b1.z; u[7]=b1.w;
                v16bf bf = __builtin_bit_cast(v16bf, u);
                #pragma unroll
                for (int mt = 0; mt < 2; ++mt) {
                    acc[mt][nt] = __builtin_amdgcn_wmma_f32_16x16x32_bf16(
                        false, af[mt], false, bf, (short)0, acc[mt][nt], false, false);
                }
            }
        }

        // next tile's DMA must have landed before buffers swap
        if (wave == 0) __builtin_amdgcn_s_wait_tensorcnt(0);
        __syncthreads();
    }

    // ---- Epilogue ----
    // A double-buffer is dead now: reuse its 32 KB as the 128x128 bf16 out tile.
    unsigned short* lOut = &lA[0][0];
    const bool do_stats = (ssum != nullptr);
    #pragma unroll
    for (int nt = 0; nt < 4; ++nt) {
        const int coln = wn + nt * 16 + lr;     // local col
        const float bv = bias[n0 + coln];
        float s = 0.f, s2 = 0.f;
        #pragma unroll
        for (int mt = 0; mt < 2; ++mt) {
            const int rloc = wm + mt * 16 + hf * 8;
            v8f a = acc[mt][nt];
            #pragma unroll
            for (int i = 0; i < 8; ++i) {
                float v = a[i] + bv;
                v = v > 0.f ? v : 0.f;
                lOut[(rloc + i) * 128 + coln] = f2bf(v);
                if (m0 + rloc + i < M) { s += v; s2 += v * v; }
            }
        }
        if (do_stats) { atomicAdd(&lcs[coln], s); atomicAdd(&lcss[coln], s2); }
    }
    __syncthreads();
    // coalesced streaming store: 2 threads per row, 8 x b128 each
    {
        const int r = t >> 1;
        const int part = (t & 1) * 64;          // ushort offset
        const int grow = m0 + r;
        if (grow < M) {
            const uint4* src = (const uint4*)(lOut + r * 128 + part);
            uint4* dst = (uint4*)(Out + (size_t)grow * DF + n0 + part);
            #pragma unroll
            for (int i = 0; i < 8; ++i) dst[i] = src[i];
        }
    }
    if (do_stats) {
        if (t < 128) {
            atomicAdd(&ssum[n0 + t], lcs[t]);
            atomicAdd(&sss[n0 + t],  lcss[t]);
        }
    }
}

// ---------------------------------------------------------------------------
// Kernel 4: global_add_pool of r3 (bf16).  batch is sorted, so each
// (64-node x 256-col) block keeps a running per-column sum in registers and
// only touches global atomics at segment boundaries (~64x fewer atomics).
// grid = (ceil(M/64), 2), block = 256.
// ---------------------------------------------------------------------------
__global__ void __launch_bounds__(256)
pool_kernel(const unsigned short* __restrict__ r3, const int* __restrict__ batch,
            float* __restrict__ psum, float* __restrict__ cnt, int M)
{
    const int t  = threadIdx.x;
    const int c  = blockIdx.y * 256 + t;
    const int n0 = blockIdx.x * 64;
    const int nend = (n0 + 64 < M) ? (n0 + 64) : M;
    int gcur = batch[n0];
    float acc = 0.f;
    float cacc = 0.f;
    const bool do_cnt = (blockIdx.y == 0) && (t == 0);
    for (int n = n0; n < nend; ++n) {
        int g = batch[n];
        if (g != gcur) {
            atomicAdd(&psum[gcur * DF + c], acc);
            if (do_cnt) atomicAdd(&cnt[gcur], cacc);
            acc = 0.f; cacc = 0.f; gcur = g;
        }
        acc += bf2f(r3[(size_t)n * DF + c]);
        cacc += 1.f;
    }
    atomicAdd(&psum[gcur * DF + c], acc);
    if (do_cnt) atomicAdd(&cnt[gcur], cacc);
}

// ---------------------------------------------------------------------------
// Kernel 5: head — BN3-on-pooled sums, fc1/prelu, fc2/sigmoid, fc3.
// ---------------------------------------------------------------------------
__global__ void __launch_bounds__(256)
head_kernel(const float* __restrict__ psum, const float* __restrict__ cnt,
            const float* __restrict__ s3sum, const float* __restrict__ s3ss,
            const float* __restrict__ bn3g, const float* __restrict__ bn3b,
            const float* __restrict__ fc1w, const float* __restrict__ fc1b,
            const float* __restrict__ fc2w, const float* __restrict__ fc2b,
            const float* __restrict__ fc3w, const float* __restrict__ fc3b,
            const float* __restrict__ a3,  float* __restrict__ out, float invN)
{
    __shared__ float h[DF], a1[DF], a2[DF / 2];
    const int g = blockIdx.x;
    const int t = threadIdx.x;
    const float c = cnt[g];
    for (int cc = t; cc < DF; cc += 256) {
        float mean = s3sum[cc] * invN;
        float var  = s3ss[cc] * invN - mean * mean;
        float sc = bn3g[cc] * rsqrtf(var + EPSBN);
        float sh = bn3b[cc] - mean * sc;
        h[cc] = psum[g * DF + cc] * sc + c * sh;
    }
    __syncthreads();
    const float alpha = a3[0];
    for (int cc = t; cc < DF; cc += 256) {
        float acc = fc1b[cc];
        for (int k = 0; k < DF; ++k) acc += h[k] * fc1w[k * DF + cc];
        a1[cc] = acc >= 0.f ? acc : alpha * acc;
    }
    __syncthreads();
    {
        float acc = fc2b[t];
        for (int k = 0; k < DF; ++k) acc += a1[k] * fc2w[k * (DF/2) + t];
        a2[t] = 1.f / (1.f + expf(-acc));
    }
    __syncthreads();
    if (t < NC) {
        float acc = fc3b[t];
        for (int k = 0; k < DF/2; ++k) acc += a2[k] * fc3w[k * NC + t];
        out[g * NC + t] = acc;
    }
}

// ---------------------------------------------------------------------------
extern "C" void kernel_launch(void* const* d_in, const int* in_sizes, int n_in,
                              void* d_out, int out_size, void* d_ws, size_t ws_size,
                              hipStream_t stream)
{
    const float* x     = (const float*)d_in[0];
    // d_in[1] edge_index: unused (ChebConv K=1)
    const int*   batch = (const int*)  d_in[2];
    const float* bn1_g = (const float*)d_in[3];
    const float* bn1_b = (const float*)d_in[4];
    const float* w0    = (const float*)d_in[5];
    const float* b0    = (const float*)d_in[6];
    const float* w1    = (const float*)d_in[7];
    const float* bb1   = (const float*)d_in[8];
    const float* w2    = (const float*)d_in[9];
    const float* bb2   = (const float*)d_in[10];
    const float* w3    = (const float*)d_in[11];
    const float* bb3   = (const float*)d_in[12];
    const float* bn3_g = (const float*)d_in[13];
    const float* bn3_b = (const float*)d_in[14];
    const float* a3    = (const float*)d_in[15];
    const float* fc1w  = (const float*)d_in[16];
    const float* fc1b  = (const float*)d_in[17];
    const float* fc2w  = (const float*)d_in[18];
    const float* fc2b  = (const float*)d_in[19];
    const float* fc3w  = (const float*)d_in[20];
    const float* fc3b  = (const float*)d_in[21];
    float* out = (float*)d_out;

    char* ws = (char*)d_ws;
    // ---- zero-initialized stats/pool region (single memset) ----
    float* sx_sum = (float*)(ws + 0);           // 128
    float* sx_ss  = (float*)(ws + 512);         // 128
    float* s1_sum = (float*)(ws + 1024);        // 512
    float* s1_ss  = (float*)(ws + 3072);
    float* s2_sum = (float*)(ws + 5120);
    float* s2_ss  = (float*)(ws + 7168);
    float* s3_sum = (float*)(ws + 9216);
    float* s3_ss  = (float*)(ws + 11264);
    float* psum   = (float*)(ws + 13312);       // 64*512
    float* cnt    = (float*)(ws + 144384);      // 64
    const size_t ZERO_BYTES = 144640;
    // ---- write-only regions ----
    float* bias0p = (float*)(ws + 144896);
    float* bias2p = (float*)(ws + 146944);
    float* bias3p = (float*)(ws + 148992);
    unsigned short* wpk0 = (unsigned short*)(ws + 151040);      // 512*128
    unsigned short* wpk1 = (unsigned short*)(ws + 282112);      // 512*512
    unsigned short* wpk2 = (unsigned short*)(ws + 806400);
    unsigned short* wpk3 = (unsigned short*)(ws + 1330688);
    unsigned short* xbf  = (unsigned short*)(ws + 1854976);     // N*128 bf16
    unsigned short* bufA = (unsigned short*)(ws + 27454976);    // N*512 bf16
    unsigned short* bufB = (unsigned short*)(ws + 129854976);   // N*512 bf16
    const size_t NEEDED = 232254976;
    if (ws_size < NEEDED) return;

    const int M = NNODES;
    const float invN = 1.0f / (float)M;
    const int mBlocks = (M + 127) / 128;

    hipMemsetAsync(ws, 0, ZERO_BYTES, stream);

    // 1. BN1 stats of x + bf16 conversion
    stats_cvt_kernel<<<mBlocks, 256, 0, stream>>>(x, xbf, sx_sum, sx_ss, M);

    // 2. weight prep (BN folds + WMMA-B packing)
    prep_w0_kernel<<<DF, 128, 0, stream>>>(w0, b0, bn1_g, bn1_b, sx_sum, sx_ss,
                                           wpk0, bias0p, invN);
    prep_plain_kernel<<<DF, 256, 0, stream>>>(w1, wpk1);

    dim3 gg(mBlocks, DF / 128);
    // 3. r0 = relu(BN1(x) @ w0 + b0)          (no stats needed)
    gemm_bf16_kernel<<<gg, 256, 0, stream>>>(xbf, wpk0, bias0p, bufA, M, INF,
                                             nullptr, nullptr);
    // 4. r1 = relu(r0 @ w1 + bb1)  + stats(r1)
    gemm_bf16_kernel<<<gg, 256, 0, stream>>>(bufA, wpk1, bb1, bufB, M, DF,
                                             s1_sum, s1_ss);
    // 5. fold BN3(stats r1) into w2; r2 = relu(BN3(r1) @ w2 + bb2) + stats(r2)
    prep_bn_kernel<<<DF, 256, 0, stream>>>(w2, bb2, bn3_g, bn3_b, s1_sum, s1_ss,
                                           wpk2, bias2p, invN);
    gemm_bf16_kernel<<<gg, 256, 0, stream>>>(bufB, wpk2, bias2p, bufA, M, DF,
                                             s2_sum, s2_ss);
    // 6. fold BN3(stats r2) into w3; r3 = relu(BN3(r2) @ w3 + bb3) + stats(r3)
    prep_bn_kernel<<<DF, 256, 0, stream>>>(w3, bb3, bn3_g, bn3_b, s2_sum, s2_ss,
                                           wpk3, bias3p, invN);
    gemm_bf16_kernel<<<gg, 256, 0, stream>>>(bufA, wpk3, bias3p, bufB, M, DF,
                                             s3_sum, s3_ss);

    // 7. pool raw r3 per graph (BN3 affine applied post-pool in head)
    dim3 pg((M + 63) / 64, 2);
    pool_kernel<<<pg, 256, 0, stream>>>(bufB, batch, psum, cnt, M);

    // 8. head: BN3-on-pooled + fc1/prelu + fc2/sigmoid + fc3
    head_kernel<<<NG, 256, 0, stream>>>(psum, cnt, s3_sum, s3_ss, bn3_g, bn3_b,
                                        fc1w, fc1b, fc2w, fc2b, fc3w, fc3b,
                                        a3, out, invN);
}
